// MetaLearner_27453430956699
// MI455X (gfx1250) — matrix operand order repaired and checked by reading, not compiled, not run on hardware
//
#include <hip/hip_runtime.h>

typedef _Float16 v4h  __attribute__((ext_vector_type(4)));
typedef _Float16 v8h  __attribute__((ext_vector_type(8)));
typedef _Float16 v16h __attribute__((ext_vector_type(16)));
typedef float    v4f  __attribute__((ext_vector_type(4)));
typedef float    v8f  __attribute__((ext_vector_type(8)));

#define BATCH 32768
#define HDIM  1024
#define NFAM  10

#define BM 64          // rows per block
#define BN 128         // column chunk
#define BK 32          // K step (WMMA K)
#define LDA 40         // padded A-tile stride in halfs (80B: 16B aligned, bank-safe)
#define LDB 40         // padded B-tile stride in halfs
#define LDH 1032       // padded hidden stride in halfs (2064B)

// ---- CDNA5 async global->LDS copy (ASYNCcnt) ------------------------------
__device__ __forceinline__ void async_load_b64(unsigned lds_off, const _Float16* g) {
    asm volatile("global_load_async_to_lds_b64 %0, %1, off"
                 :: "v"(lds_off), "v"(g) : "memory");
}
template <int N>
__device__ __forceinline__ void wait_async() {
    asm volatile("s_wait_asynccnt %0" :: "i"(N) : "memory");
}

// ---------------------------------------------------------------------------
// Kernel 0: Wt[n][k] = (f16) W[k][n]   (one-time f32->f16 transpose)
// ---------------------------------------------------------------------------
__global__ __launch_bounds__(256) void transpose_to_f16_kernel(
    const float* __restrict__ W, _Float16* __restrict__ Wt, int K, int N)
{
    __shared__ float t[32][33];
    const int tid = threadIdx.x;
    const int ntn = N >> 5;
    const int bn = (blockIdx.x % ntn) << 5;
    const int bk = (blockIdx.x / ntn) << 5;
    {
        const int c = tid & 31, r0 = (tid >> 5) * 4;
#pragma unroll
        for (int i = 0; i < 4; ++i)
            t[r0 + i][c] = W[(size_t)(bk + r0 + i) * N + bn + c];
    }
    __syncthreads();
    {
        const int r = tid >> 3, c4 = (tid & 7) * 4;
        v4h h;
#pragma unroll
        for (int j = 0; j < 4; ++j) h[j] = (_Float16)t[c4 + j][r];
        *(v4h*)&Wt[(size_t)(bn + r) * K + bk + c4] = h;
    }
}

// ---------------------------------------------------------------------------
// Kernel 1: nearest prototype per row.  argmin_j ( p2[j] - 2 * f . p_j )
// ---------------------------------------------------------------------------
__global__ __launch_bounds__(256) void nearest_kernel(
    const float* __restrict__ F, const float* __restrict__ P,
    int* __restrict__ nidx)
{
    __shared__ float sP[NFAM * HDIM];
    __shared__ float sP2[NFAM];
    const int tid = threadIdx.x;

    for (int i = tid; i < NFAM * HDIM; i += 256) sP[i] = P[i];
    __syncthreads();
    if (tid < NFAM) {
        float s = 0.f;
        for (int k = 0; k < HDIM; ++k) { float v = sP[tid * HDIM + k]; s += v * v; }
        sP2[tid] = s;
    }
    __syncthreads();

    const int wid = tid >> 5, lane = tid & 31;
    for (int r = 0; r < 8; ++r) {
        const int row = blockIdx.x * 64 + wid * 8 + r;
        const float* f = F + (size_t)row * HDIM;
        float acc[NFAM];
#pragma unroll
        for (int j = 0; j < NFAM; ++j) acc[j] = 0.f;
        for (int k = lane; k < HDIM; k += 32) {
            const float fv = f[k];
#pragma unroll
            for (int j = 0; j < NFAM; ++j) acc[j] += fv * sP[j * HDIM + k];
        }
#pragma unroll
        for (int j = 0; j < NFAM; ++j) {
            float v = acc[j];
            for (int m = 16; m > 0; m >>= 1) v += __shfl_xor(v, m, 32);
            acc[j] = v;
        }
        if (lane == 0) {
            int best = 0;
            float bd = sP2[0] - 2.f * acc[0];
#pragma unroll
            for (int j = 1; j < NFAM; ++j) {
                const float d = sP2[j] - 2.f * acc[j];
                if (d < bd) { bd = d; best = j; }
            }
            nidx[row] = best;
        }
    }
}

// ---------------------------------------------------------------------------
// Kernel 2: fused  hidden = relu([F, P[nid]] @ W1 + b1);  out = hidden @ W2 + b2
// W1t/W2t are f16, pre-transposed [n][k].  B tiles double-buffered via
// global_load_async_to_lds_b64.  hidden (f16) lives entirely in LDS.
// ---------------------------------------------------------------------------
__global__ __launch_bounds__(256) void mlp_kernel(
    const float* __restrict__ F,      const float* __restrict__ P,
    const _Float16* __restrict__ W1t, const float* __restrict__ b1,
    const _Float16* __restrict__ W2t, const float* __restrict__ b2,
    const int* __restrict__ nidx,     float* __restrict__ out)
{
    __shared__ _Float16 sH[BM * LDH];     // 129 KB hidden tile
    __shared__ _Float16 sA[BM * LDA];     // 5 KB
    __shared__ _Float16 sB[2][BN * LDB];  // 2 x 10 KB (double buffered)

    const int tid  = threadIdx.x;
    const int wid  = tid >> 5, lane = tid & 31;
    const int wm   = wid >> 2;            // rows 32*wm
    const int wn   = wid & 3;             // cols 32*wn
    const int row0 = blockIdx.x * BM;

    const int fm = lane & 15;
    const int kh = lane >> 4;

    // ---- per-thread A-fill pointers (2 rows, 4 consecutive floats each) ----
    const int rA = tid >> 3;              // 0..31
    const int cA = (tid & 7) * 4;
    const float* fRow0 = F + (size_t)(row0 + rA) * HDIM + cA;
    const float* fRow1 = F + (size_t)(row0 + rA + 32) * HDIM + cA;
    const float* pRow0 = P + (size_t)nidx[row0 + rA] * HDIM + cA;
    const float* pRow1 = P + (size_t)nidx[row0 + rA + 32] * HDIM + cA;
    _Float16* sA0 = &sA[rA * LDA + cA];
    _Float16* sA1 = &sA[(rA + 32) * LDA + cA];

    // ---- per-thread async B-fill addressing (4 x b64 per tile) ------------
    const int rB = tid >> 3;              // n row 0..31 (+32*i)
    const int cB = (tid & 7) * 4;         // k offset in halfs
    unsigned ldsB0 = (unsigned)(uintptr_t)&sB[0][rB * LDB + cB];
    unsigned ldsB1 = (unsigned)(uintptr_t)&sB[1][rB * LDB + cB];

    auto issueB = [&](const _Float16* Wt, int LDW, int n0, int k0, int buf) {
        const unsigned base = buf ? ldsB1 : ldsB0;
        const _Float16* g = Wt + (size_t)(n0 + rB) * LDW + k0 + cB;
#pragma unroll
        for (int i = 0; i < 4; ++i)
            async_load_b64(base + (unsigned)(i * 32 * LDB * 2),
                           g + (size_t)(32 * i) * LDW);
    };

    auto loadFragA = [&](const _Float16* ap) {
        v8h lo = *(const v8h*)ap;
        v8h hi = *(const v8h*)(ap + 16);
        return __builtin_shufflevector(lo, hi, 0,1,2,3,4,5,6,7,8,9,10,11,12,13,14,15);
    };
    auto loadFragB = [&](const _Float16* bp) {
        v8h lo = *(const v8h*)bp;
        v8h hi = *(const v8h*)(bp + 16);
        return __builtin_shufflevector(lo, hi, 0,1,2,3,4,5,6,7,8,9,10,11,12,13,14,15);
    };

    // ======================= GEMM1 : K = 2H ==============================
    for (int nc = 0; nc < HDIM / BN; ++nc) {
        const int n0 = nc * BN;
        v8f acc[2][2];
#pragma unroll
        for (int tm = 0; tm < 2; ++tm)
#pragma unroll
            for (int tn = 0; tn < 2; ++tn)
#pragma unroll
                for (int v = 0; v < 8; ++v) acc[tm][tn][v] = 0.f;

        issueB(W1t, 2 * HDIM, n0, 0, 0);

        for (int ks = 0; ks < (2 * HDIM) / BK; ++ks) {
            const int k0 = ks * BK;
            __syncthreads();                       // prev reads of sA/sB done
            // fill A tile (f32 -> f16), features then nearest prototypes
            {
                const float* s0 = (ks < 32) ? (fRow0 + k0) : (pRow0 + (k0 - HDIM));
                const float* s1 = (ks < 32) ? (fRow1 + k0) : (pRow1 + (k0 - HDIM));
                v4f va = *(const v4f*)s0;
                v4f vb = *(const v4f*)s1;
                v4h ha, hb;
#pragma unroll
                for (int j = 0; j < 4; ++j) { ha[j] = (_Float16)va[j]; hb[j] = (_Float16)vb[j]; }
                *(v4h*)sA0 = ha;
                *(v4h*)sA1 = hb;
            }
            if (ks + 1 < (2 * HDIM) / BK) {
                issueB(W1t, 2 * HDIM, n0, k0 + BK, (ks + 1) & 1);
                wait_async<4>();                   // current buffer complete
            } else {
                wait_async<0>();
            }
            __syncthreads();                       // A + B(ks) visible to all

            const _Float16* bufB = sB[ks & 1];
            v16h a[2], b[2];
#pragma unroll
            for (int t = 0; t < 2; ++t) {
                a[t] = loadFragA(&sA[(32 * wm + 16 * t + fm) * LDA + kh * 8]);
                b[t] = loadFragB(&bufB[(32 * wn + 16 * t + fm) * LDB + kh * 8]);
            }
#pragma unroll
            for (int tm = 0; tm < 2; ++tm)
#pragma unroll
                for (int tn = 0; tn < 2; ++tn)
                    acc[tm][tn] = __builtin_amdgcn_wmma_f32_16x16x32_f16(
                        false, a[tm], false, b[tn], (short)0, acc[tm][tn], false, false);
        }

        // epilogue: bias + relu -> hidden f16 in LDS
#pragma unroll
        for (int tm = 0; tm < 2; ++tm)
#pragma unroll
            for (int tn = 0; tn < 2; ++tn) {
                const int n  = n0 + 32 * wn + 16 * tn + fm;
                const float bv = b1[n];
                const int mb = 32 * wm + 16 * tm + 8 * kh;
#pragma unroll
                for (int v = 0; v < 8; ++v) {
                    float hv = acc[tm][tn][v] + bv;
                    hv = hv > 0.f ? hv : 0.f;
                    sH[(mb + v) * LDH + n] = (_Float16)hv;
                }
            }
    }

    // ======================= GEMM2 : K = H ===============================
    for (int nc = 0; nc < HDIM / BN; ++nc) {
        const int n0 = nc * BN;
        v8f acc[2][2];
#pragma unroll
        for (int tm = 0; tm < 2; ++tm)
#pragma unroll
            for (int tn = 0; tn < 2; ++tn)
#pragma unroll
                for (int v = 0; v < 8; ++v) acc[tm][tn][v] = 0.f;

        issueB(W2t, HDIM, n0, 0, 0);

        for (int ks = 0; ks < HDIM / BK; ++ks) {
            const int k0 = ks * BK;
            __syncthreads();
            if (ks + 1 < HDIM / BK) {
                issueB(W2t, HDIM, n0, k0 + BK, (ks + 1) & 1);
                wait_async<4>();
            } else {
                wait_async<0>();
            }
            __syncthreads();

            const _Float16* bufB = sB[ks & 1];
            v16h a[2], b[2];
#pragma unroll
            for (int t = 0; t < 2; ++t) {
                a[t] = loadFragA(&sH[(32 * wm + 16 * t + fm) * LDH + k0 + kh * 8]);
                b[t] = loadFragB(&bufB[(32 * wn + 16 * t + fm) * LDB + kh * 8]);
            }
#pragma unroll
            for (int tm = 0; tm < 2; ++tm)
#pragma unroll
                for (int tn = 0; tn < 2; ++tn)
                    acc[tm][tn] = __builtin_amdgcn_wmma_f32_16x16x32_f16(
                        false, a[tm], false, b[tn], (short)0, acc[tm][tn], false, false);
        }

        // epilogue: + b2, store f32
#pragma unroll
        for (int tm = 0; tm < 2; ++tm)
#pragma unroll
            for (int tn = 0; tn < 2; ++tn) {
                const int n  = n0 + 32 * wn + 16 * tn + fm;
                const float bv = b2[n];
                const int mb = 32 * wm + 16 * tm + 8 * kh;
#pragma unroll
                for (int v = 0; v < 8; ++v)
                    out[(size_t)(row0 + mb + v) * HDIM + n] = acc[tm][tn][v] + bv;
            }
    }
}

extern "C" void kernel_launch(void* const* d_in, const int* in_sizes, int n_in,
                              void* d_out, int out_size, void* d_ws, size_t ws_size,
                              hipStream_t stream) {
    const float* F  = (const float*)d_in[0];
    const float* P  = (const float*)d_in[1];
    const float* W1 = (const float*)d_in[2];
    const float* b1 = (const float*)d_in[3];
    const float* W2 = (const float*)d_in[4];
    const float* b2 = (const float*)d_in[5];
    float* out = (float*)d_out;

    int*      nid = (int*)d_ws;                                   // 128 KB
    _Float16* W1t = (_Float16*)((char*)d_ws + (size_t)BATCH * 4); // 4 MB  [1024][2048]
    _Float16* W2t = W1t + (size_t)2 * HDIM * HDIM;                // 2 MB  [1024][1024]

    nearest_kernel<<<BATCH / 64, 256, 0, stream>>>(F, P, nid);
    transpose_to_f16_kernel<<<(2 * HDIM / 32) * (HDIM / 32), 256, 0, stream>>>(W1, W1t, 2 * HDIM, HDIM);
    transpose_to_f16_kernel<<<(HDIM / 32) * (HDIM / 32), 256, 0, stream>>>(W2, W2t, HDIM, HDIM);
    mlp_kernel<<<BATCH / BM, 256, 0, stream>>>(F, P, W1t, b1, W2t, b2, nid, out);
}
